// MemoryAugmentedModel_14559939133709
// MI455X (gfx1250) — matrix operand order repaired and checked by reading, not compiled, not run on hardware
//
#include <hip/hip_runtime.h>
#include <stddef.h>

// ---------------- problem constants ----------------
#define VOCAB   512
#define MEM_DIM 64
#define QDIM    32
#define NUM_MEM (VOCAB * VOCAB)   // 262144
#define BATCH   1024

#define NSPLIT1 64
#define CHUNK1  (NUM_MEM / NSPLIT1)   // 4096
#define NSPLIT2 128
#define CHUNK2  (NUM_MEM / NSPLIT2)   // 2048

typedef __attribute__((ext_vector_type(16))) __bf16 v16bf;
typedef __attribute__((ext_vector_type(8)))  __bf16 v8bf;
typedef __attribute__((ext_vector_type(8)))  float  v8f;
typedef __attribute__((ext_vector_type(4)))  float  v4f;

// float -> bf16, round-to-nearest-even
__device__ __forceinline__ __bf16 f2bf(float f) {
    union { float f; unsigned u; } v; v.f = f;
    unsigned r = v.u + 0x7fffu + ((v.u >> 16) & 1u);
    unsigned short h = (unsigned short)(r >> 16);
    __bf16 b; __builtin_memcpy(&b, &h, 2);
    return b;
}

__device__ __forceinline__ v16bf mk16(v8bf lo, v8bf hi) {
    v16bf r;
#pragma unroll
    for (int i = 0; i < 8; ++i) { r[i] = lo[i]; r[i + 8] = hi[i]; }
    return r;
}

__device__ __forceinline__ v8f wmma_bf16(v16bf a, v16bf b, v8f c) {
    return __builtin_amdgcn_wmma_f32_16x16x32_bf16(false, a, false, b, (short)0, c, false, false);
}

// A-matrix (16x32 bf16) slice for this lane from row-major [.,64] bf16 storage.
__device__ __forceinline__ v16bf load_qA(const __bf16* q, int rowbase, int kstep, int lane) {
    const __bf16* rp = q + (size_t)(rowbase + (lane & 15)) * MEM_DIM
                         + kstep * 32 + ((lane < 16) ? 0 : 8);
    v8bf lo = *(const v8bf*)rp;
    v8bf hi = *(const v8bf*)(rp + 16);
    return mk16(lo, hi);
}

// B-tile from the pre-packed bf16 layout: tile = 1 KB = 32 lanes x 32 B.
__device__ __forceinline__ v16bf load_packedB(const __bf16* base, size_t tile, int lane) {
    const __bf16* p = base + tile * 512 + (size_t)lane * 8;
    v8bf lo = *(const v8bf*)p;
    v8bf hi = *(const v8bf*)(p + 256);
    return mk16(lo, hi);
}

// Fallback: B-tile (32x16) = keys^T gathered from f32 + converted in-register.
__device__ __forceinline__ v16bf load_keysB_f32(const float* keys, int nbase, int dbase, int lane) {
    const float* rp = keys + (size_t)(nbase + (lane & 15)) * MEM_DIM
                           + dbase + ((lane < 16) ? 0 : 8);
    v16bf b;
#pragma unroll
    for (int i = 0; i < 4; ++i) {
        v4f f = *(const v4f*)(rp + (i & 1) * 4 + (i >> 1) * 16);
        b[i * 4 + 0] = f2bf(f[0]); b[i * 4 + 1] = f2bf(f[1]);
        b[i * 4 + 2] = f2bf(f[2]); b[i * 4 + 3] = f2bf(f[3]);
    }
    return b;
}

// ---------------- kernel 1: query MLP -> q in bf16 ----------------
__global__ __launch_bounds__(256)
void mlp_kernel(const int* __restrict__ x,
                const float* __restrict__ W1, const float* __restrict__ b1,
                const float* __restrict__ W2, const float* __restrict__ b2,
                const float* __restrict__ W3, const float* __restrict__ b3,
                unsigned short* __restrict__ q_raw) {
    int b = blockIdx.x * blockDim.x + threadIdx.x;
    if (b >= BATCH) return;
    __bf16* qb = (__bf16*)q_raw;
    float x0 = (float)x[b * 2 + 0];
    float x1 = (float)x[b * 2 + 1];
    float h1[QDIM];
#pragma unroll
    for (int j = 0; j < QDIM; ++j) {
        float s = fmaf(W1[j * 2], x0, fmaf(W1[j * 2 + 1], x1, b1[j]));
        h1[j] = fmaxf(s, 0.0f);
    }
    float h2[QDIM];
    for (int j = 0; j < QDIM; ++j) {
        float s = b2[j];
#pragma unroll
        for (int k = 0; k < QDIM; ++k) s = fmaf(W2[j * QDIM + k], h1[k], s);
        h2[j] = fmaxf(s, 0.0f);
    }
    for (int j = 0; j < MEM_DIM; ++j) {
        float s = b3[j];
#pragma unroll
        for (int k = 0; k < QDIM; ++k) s = fmaf(W3[j * QDIM + k], h2[k], s);
        qb[(size_t)b * MEM_DIM + j] = f2bf(s);
    }
}

// ------------- pack kernels: f32 -> bf16 in exact WMMA-B lane layout -------------
__global__ __launch_bounds__(256)
void pack_keys_kernel(const float* __restrict__ keys, unsigned short* __restrict__ out_raw) {
    __bf16* outp = (__bf16*)out_raw;
    int t = blockIdx.x * 256 + threadIdx.x;          // 16-byte group id
    int tile = t >> 6;
    int r    = t & 63;
    int lane = r & 31;
    int g    = r >> 5;
    int nt   = tile >> 1;
    int dblk = tile & 1;
    int n    = nt * 16 + (lane & 15);
    int d    = dblk * 32 + ((lane & 16) ? 8 : 0) + g * 16;
    const float* src = keys + (size_t)n * MEM_DIM + d;
    v4f a = *(const v4f*)src;
    v4f b = *(const v4f*)(src + 4);
    v8bf o;
#pragma unroll
    for (int i = 0; i < 4; ++i) { o[i] = f2bf(a[i]); o[i + 4] = f2bf(b[i]); }
    *(v8bf*)(outp + (size_t)t * 8) = o;
}

__global__ __launch_bounds__(256)
void pack_vals_kernel(const float* __restrict__ vals, unsigned short* __restrict__ out_raw) {
    __bf16* outp = (__bf16*)out_raw;
    int t = blockIdx.x * 256 + threadIdx.x;
    int tile = t >> 6;
    int r    = t & 63;
    int lane = r & 31;
    int g    = r >> 5;
    int kb2  = tile >> 5;
    int nt   = tile & 31;
    int n    = nt * 16 + (lane & 15);
    int kbase = kb2 * 32 + ((lane & 16) ? 8 : 0) + g * 16;
    const float* src = vals + (size_t)kbase * VOCAB + n;
    v8bf o;
#pragma unroll
    for (int i = 0; i < 8; ++i) o[i] = f2bf(src[(size_t)i * VOCAB]);
    *(v8bf*)(outp + (size_t)t * 8) = o;
}

// ---------------- kernel 2: per-chunk softmax stats, two-phase ----------------
// grid = (NSPLIT1, BATCH/128), block = 256 (8 waves, wave w owns rows blk.y*128+w*16..+15)
// Phase A: chunk max (no exp). Phase B: sum exp(s-m). Halves trans-op count vs online.
template <bool KP>
__global__ __launch_bounds__(256)
void stats_kernel(const unsigned short* __restrict__ q_raw,
                  const float* __restrict__ keys,
                  const unsigned short* __restrict__ keysB_raw,
                  float* __restrict__ Mpart, float* __restrict__ Lpart) {
    const __bf16* qb    = (const __bf16*)q_raw;
    const __bf16* keysB = (const __bf16*)keysB_raw;
    int lane = threadIdx.x & 31;
    int w    = threadIdx.x >> 5;
    int mtb  = blockIdx.y * 128 + w * 16;

    v16bf A0 = load_qA(qb, mtb, 0, lane);
    v16bf A1 = load_qA(qb, mtb, 1, lane);

    const v8f z = {0.f, 0.f, 0.f, 0.f, 0.f, 0.f, 0.f, 0.f};
    int cbase = blockIdx.x * CHUNK1;

    float m[8];
#pragma unroll
    for (int i = 0; i < 8; ++i) m[i] = -3.0e38f;

    // ---- phase A: row maxima over the chunk ----
    for (int n = cbase; n < cbase + CHUNK1; n += 16) {
        v16bf B0, B1;
        if constexpr (KP) {
            size_t nt = (size_t)(n >> 4);
            B0 = load_packedB(keysB, nt * 2 + 0, lane);
            B1 = load_packedB(keysB, nt * 2 + 1, lane);
        } else {
            B0 = load_keysB_f32(keys, n, 0, lane);
            B1 = load_keysB_f32(keys, n, 32, lane);
        }
        v8f S = wmma_bf16(A0, B0, z);
        S = wmma_bf16(A1, B1, S);
#pragma unroll
        for (int i = 0; i < 8; ++i) m[i] = fmaxf(m[i], S[i]);
    }
#pragma unroll
    for (int i = 0; i < 8; ++i)          // butterfly max across the 16-lane column group
        for (int off = 1; off < 16; off <<= 1)
            m[i] = fmaxf(m[i], __shfl_xor(m[i], off, 32));

    // ---- phase B: sum of exp(s - m) over the chunk ----
    float l[8];
#pragma unroll
    for (int i = 0; i < 8; ++i) l[i] = 0.0f;
    for (int n = cbase; n < cbase + CHUNK1; n += 16) {
        v16bf B0, B1;
        if constexpr (KP) {
            size_t nt = (size_t)(n >> 4);
            B0 = load_packedB(keysB, nt * 2 + 0, lane);
            B1 = load_packedB(keysB, nt * 2 + 1, lane);
        } else {
            B0 = load_keysB_f32(keys, n, 0, lane);
            B1 = load_keysB_f32(keys, n, 32, lane);
        }
        v8f S = wmma_bf16(A0, B0, z);
        S = wmma_bf16(A1, B1, S);
#pragma unroll
        for (int i = 0; i < 8; ++i) l[i] += __expf(S[i] - m[i]);
    }
#pragma unroll
    for (int i = 0; i < 8; ++i)          // butterfly sum
        for (int off = 1; off < 16; off <<= 1)
            l[i] += __shfl_xor(l[i], off, 32);

    if ((lane & 15) == 0) {
        int rb = mtb + ((lane < 16) ? 0 : 8);
#pragma unroll
        for (int i = 0; i < 8; ++i) {
            Mpart[(size_t)blockIdx.x * BATCH + rb + i] = m[i];
            Lpart[(size_t)blockIdx.x * BATCH + rb + i] = l[i];
        }
    }
}

// ---------------- kernel 3: merge chunk stats -> m, 1/l ----------------
__global__ __launch_bounds__(256)
void stats_reduce_kernel(const float* __restrict__ Mpart, const float* __restrict__ Lpart,
                         float* __restrict__ Mfin, float* __restrict__ Linv) {
    int b = blockIdx.x * blockDim.x + threadIdx.x;
    if (b >= BATCH) return;
    float m = -3.0e38f;
    for (int s = 0; s < NSPLIT1; ++s) m = fmaxf(m, Mpart[(size_t)s * BATCH + b]);
    float l = 0.0f;
    for (int s = 0; s < NSPLIT1; ++s)
        l += Lpart[(size_t)s * BATCH + b] * __expf(Mpart[(size_t)s * BATCH + b] - m);
    Mfin[b] = m;
    Linv[b] = 1.0f / l;
}

// ---------------- kernel 4: fused softmax chunk + P @ V (flash pass 2) ---------
// grid = (NSPLIT2, BATCH/128), block = 512 (16 waves):
//   rt = w&7 -> 16-row tile, ch = w>>3 -> 256-column half.
// 128 batch rows per WG halves packed-V L2 replication (8x instead of 16x).
template <bool KP, bool VP>
__global__ __launch_bounds__(512)
void out_kernel(const unsigned short* __restrict__ q_raw,
                const float* __restrict__ keys,
                const unsigned short* __restrict__ keysB_raw,
                const float* __restrict__ vals,
                const unsigned short* __restrict__ valsB_raw,
                const float* __restrict__ Mfin, const float* __restrict__ Linv,
                float* __restrict__ out) {
    __shared__ __attribute__((aligned(16))) __bf16 Pl[8][16][40];  // bank-padded

    const __bf16* qb    = (const __bf16*)q_raw;
    const __bf16* keysB = (const __bf16*)keysB_raw;
    const __bf16* valsB = (const __bf16*)valsB_raw;
    int lane = threadIdx.x & 31;
    int w    = threadIdx.x >> 5;
    int rt   = w & 7;
    int ch   = w >> 3;
    int rowbase = blockIdx.y * 128 + rt * 16;

    v16bf A0 = load_qA(qb, rowbase, 0, lane);
    v16bf A1 = load_qA(qb, rowbase, 1, lane);

    float mrow[8], lrow[8];
    {
        int rb = rowbase + ((lane < 16) ? 0 : 8);
#pragma unroll
        for (int i = 0; i < 8; ++i) { mrow[i] = Mfin[rb + i]; lrow[i] = Linv[rb + i]; }
    }

    const v8f z = {0.f, 0.f, 0.f, 0.f, 0.f, 0.f, 0.f, 0.f};
    v8f acc[16];
#pragma unroll
    for (int ct = 0; ct < 16; ++ct) acc[ct] = z;

    int cbase = blockIdx.x * CHUNK2;
    int cend  = cbase + CHUNK2;
    for (int kb = cbase; kb < cend; kb += 32) {
        // phase 1: 16 waves build the 8 x (16x32) probability tiles
        {
            int nb = kb + ch * 16;
            v16bf B0, B1;
            if constexpr (KP) {
                size_t nt = (size_t)(nb >> 4);
                B0 = load_packedB(keysB, nt * 2 + 0, lane);
                B1 = load_packedB(keysB, nt * 2 + 1, lane);
            } else {
                B0 = load_keysB_f32(keys, nb, 0, lane);
                B1 = load_keysB_f32(keys, nb, 32, lane);
            }
            v8f S = wmma_bf16(A0, B0, z);
            S = wmma_bf16(A1, B1, S);
            int colL = ch * 16 + (lane & 15);
            int rl   = (lane < 16) ? 0 : 8;
#pragma unroll
            for (int i = 0; i < 8; ++i) {
                float p = __expf(S[i] - mrow[i]) * lrow[i];
                Pl[rt][rl + i][colL] = f2bf(p);
            }
        }
        __syncthreads();
        // phase 2: C-layout -> A-layout via LDS, then accumulate P @ V
        {
            int row = lane & 15;
            int k1  = (lane < 16) ? 0 : 8;
            v8bf lo = *(const v8bf*)&Pl[rt][row][k1];
            v8bf hi = *(const v8bf*)&Pl[rt][row][k1 + 16];
            v16bf Ap = mk16(lo, hi);

            if constexpr (VP) {
                size_t tbase = (size_t)(kb >> 5) * 32 + ch * 16;
                if (kb + 32 < cend)
                    __builtin_prefetch(valsB + (tbase + 32) * 512 + (size_t)lane * 16, 0, 1);
#pragma unroll
                for (int ct = 0; ct < 16; ++ct) {
                    v16bf Bv = load_packedB(valsB, tbase + ct, lane);
                    acc[ct] = wmma_bf16(Ap, Bv, acc[ct]);
                }
            } else {
                int col = ch * 256 + (lane & 15);
                int k0  = (lane < 16) ? 0 : 8;
                const float* vbase = vals + (size_t)(kb + k0) * VOCAB + col;
                if (kb + 32 < cend)
                    __builtin_prefetch(vbase + (size_t)32 * VOCAB, 0, 1);
#pragma unroll
                for (int ct = 0; ct < 16; ++ct) {
                    const float* vp = vbase + ct * 16;
                    v16bf Bv;
#pragma unroll
                    for (int i = 0; i < 16; ++i) Bv[i] = f2bf(vp[(size_t)(i + (i & 8)) * VOCAB]);
                    acc[ct] = wmma_bf16(Ap, Bv, acc[ct]);
                }
            }
        }
        __syncthreads();
    }

    int rl = (lane < 16) ? 0 : 8;
#pragma unroll
    for (int ct = 0; ct < 16; ++ct) {
        int col = ch * 256 + ct * 16 + (lane & 15);
#pragma unroll
        for (int i = 0; i < 8; ++i)
            atomicAdd(&out[(size_t)(rowbase + rl + i) * VOCAB + col], acc[ct][i]);
    }
}

// ---------------- host launcher ----------------
extern "C" void kernel_launch(void* const* d_in, const int* in_sizes, int n_in,
                              void* d_out, int out_size, void* d_ws, size_t ws_size,
                              hipStream_t stream) {
    (void)in_sizes; (void)n_in;
    const int*   x    = (const int*)d_in[0];
    const float* W1   = (const float*)d_in[1];
    const float* b1   = (const float*)d_in[2];
    const float* W2   = (const float*)d_in[3];
    const float* b2   = (const float*)d_in[4];
    const float* W3   = (const float*)d_in[5];
    const float* b3   = (const float*)d_in[6];
    const float* keys = (const float*)d_in[7];   // (NUM_MEM, 64)  f32
    const float* vals = (const float*)d_in[8];   // (NUM_MEM, 512) f32
    float* out = (float*)d_out;

    const size_t OFF_Q     = 0;                    // 1024*64 bf16   = 128 KB
    const size_t OFF_MP    = 131072;               // 64*1024 f32    = 256 KB
    const size_t OFF_LP    = 393216;               // 64*1024 f32    = 256 KB
    const size_t OFF_MF    = 655360;
    const size_t OFF_LI    = 659456;
    const size_t OFF_KEYSB = 663552;               // 32 MB
    const size_t SZ_KEYSB  = (size_t)NUM_MEM * MEM_DIM * 2;
    const size_t OFF_VALSB = OFF_KEYSB + SZ_KEYSB; // 256 MB
    const size_t SZ_VALSB  = (size_t)NUM_MEM * VOCAB * 2;

    char* ws = (char*)d_ws;
    unsigned short* qb    = (unsigned short*)(ws + OFF_Q);
    float* Mpart          = (float*)(ws + OFF_MP);
    float* Lpart          = (float*)(ws + OFF_LP);
    float* Mfin           = (float*)(ws + OFF_MF);
    float* Linv           = (float*)(ws + OFF_LI);
    unsigned short* keysB = (unsigned short*)(ws + OFF_KEYSB);
    unsigned short* valsB = (unsigned short*)(ws + OFF_VALSB);

    const bool kp = ws_size >= OFF_KEYSB + SZ_KEYSB;
    const bool vp = kp && (ws_size >= OFF_VALSB + SZ_VALSB);

    hipMemsetAsync(d_out, 0, (size_t)out_size * sizeof(float), stream);

    mlp_kernel<<<dim3(BATCH / 256), dim3(256), 0, stream>>>(x, W1, b1, W2, b2, W3, b3, qb);

    if (kp) pack_keys_kernel<<<dim3(8192),  dim3(256), 0, stream>>>(keys, keysB);
    if (vp) pack_vals_kernel<<<dim3(65536), dim3(256), 0, stream>>>(vals, valsB);

    dim3 sgrid(NSPLIT1, BATCH / 128), ogrid(NSPLIT2, BATCH / 128);
    if (kp) stats_kernel<true ><<<sgrid, dim3(256), 0, stream>>>(qb, keys, keysB, Mpart, Lpart);
    else    stats_kernel<false><<<sgrid, dim3(256), 0, stream>>>(qb, keys, keysB, Mpart, Lpart);

    stats_reduce_kernel<<<dim3(BATCH / 256), dim3(256), 0, stream>>>(Mpart, Lpart, Mfin, Linv);

    if (vp)
        out_kernel<true, true ><<<ogrid, dim3(512), 0, stream>>>(qb, keys, keysB, vals, valsB, Mfin, Linv, out);
    else if (kp)
        out_kernel<true, false><<<ogrid, dim3(512), 0, stream>>>(qb, keys, keysB, vals, valsB, Mfin, Linv, out);
    else
        out_kernel<false,false><<<ogrid, dim3(512), 0, stream>>>(qb, keys, keysB, vals, valsB, Mfin, Linv, out);
}
